// MemBasic_84413287235873
// MI455X (gfx1250) — matrix operand order repaired and checked by reading, not compiled, not run on hardware
//
#include <hip/hip_runtime.h>
#include <math.h>

typedef _Float16 f16_t;
typedef __attribute__((ext_vector_type(16))) _Float16 v16h;
typedef __attribute__((ext_vector_type(4)))  _Float16 v4h;
typedef __attribute__((ext_vector_type(8)))  float    v8f;

#define MEM_SIZE 131072
#define DIM      512
#define BATCH    512
#define TOPK     32

// K2 tiling
#define BT     128                  // batch rows per block
#define MT     128                  // key rows per tile
#define KT     32                   // k per wmma step
#define TILES  8                    // key tiles per block (512-block grid)
#define MCHUNK (MT * TILES)         // 1024 keys per block
#define NBCH   (BATCH / BT)         // 4
#define NMCH   (MEM_SIZE / MCHUNK)  // 128

__device__ __forceinline__ bool better(float s, int i, float s2, int i2) {
    return (s > s2) || (s == s2 && i < i2);
}

// ---------------------------------------------------------------------------
// K1: q = x @ Wq^T + bq, row L2-normalize, emit f16 qn. One block per row.
// ---------------------------------------------------------------------------
__global__ __launch_bounds__(256) void k1_project_norm(
        const float* __restrict__ x, const float* __restrict__ Wq,
        const float* __restrict__ bq, f16_t* __restrict__ qn) {
    __shared__ __align__(16) float sx[DIM];
    __shared__ float red[256];
    const int b = blockIdx.x;
    const int t = threadIdx.x;
    sx[t]       = x[(size_t)b * DIM + t];
    sx[t + 256] = x[(size_t)b * DIM + t + 256];
    __syncthreads();
    const float4* w0 = (const float4*)(Wq + (size_t)t * DIM);
    const float4* w1 = (const float4*)(Wq + (size_t)(t + 256) * DIM);
    const float4* xv = (const float4*)sx;
    float q0 = bq[t], q1 = bq[t + 256];
#pragma unroll 4
    for (int k = 0; k < DIM / 4; ++k) {
        float4 a = xv[k];
        float4 c0 = w0[k], c1 = w1[k];
        q0 = fmaf(a.x, c0.x, q0); q0 = fmaf(a.y, c0.y, q0);
        q0 = fmaf(a.z, c0.z, q0); q0 = fmaf(a.w, c0.w, q0);
        q1 = fmaf(a.x, c1.x, q1); q1 = fmaf(a.y, c1.y, q1);
        q1 = fmaf(a.z, c1.z, q1); q1 = fmaf(a.w, c1.w, q1);
    }
    red[t] = q0 * q0 + q1 * q1;
    __syncthreads();
    for (int off = 128; off > 0; off >>= 1) {
        if (t < off) red[t] += red[t + off];
        __syncthreads();
    }
    const float scale = 1.0f / fmaxf(sqrtf(red[0]), 1e-12f);
    qn[(size_t)b * DIM + t]       = (f16_t)(q0 * scale);
    qn[(size_t)b * DIM + t + 256] = (f16_t)(q1 * scale);
}

// ---------------------------------------------------------------------------
// K2: fused WMMA cosine-score GEMM + per-chunk exact top-32.
// grid = (NBCH, NMCH), block = 256 threads (8 waves), tile 128x128.
// ---------------------------------------------------------------------------
__global__ __launch_bounds__(256) void k2_score_topk(
        const f16_t* __restrict__ qn, const float* __restrict__ keys,
        float* __restrict__ candS, int* __restrict__ candI) {
    // Aliased LDS region: [GEMM] lA(10240)+lB(10240) | [select] lScore(33280)
    //                     | [final merge] mS(16384)+mI(16384)
    __shared__ __align__(16) unsigned char smem[33280];
    __shared__ float lRnorm[MT];
    f16_t* lA     = (f16_t*)smem;             // 128 rows * 80 B (32 f16 + pad)
    f16_t* lB     = (f16_t*)(smem + 10240);   // 128 rows * 80 B
    float* lScore = (float*)smem;             // 128 * 65 f32
    float* mS     = (float*)smem;             // 128 * 32 f32
    int*   mI     = (int*)(smem + 16384);     // 128 * 32 i32

    const int t    = threadIdx.x;
    const int lane = t & 31;
    const int w    = t >> 5;
    const int wr   = w >> 1;        // 0..3  -> 32-row group
    const int wc   = w & 1;         // 0..1  -> 64-col group
    const int rowBase = wr * 32;
    const int colBase = wc * 64;
    const int ln15 = lane & 15;
    const int lhi  = lane >> 4;

    const int bch   = blockIdx.x;
    const int mch   = blockIdx.y;
    const int bRow0 = bch * BT;

    // Running top-32 (sorted desc) in registers; 2 threads per batch row.
    float tS[TOPK];
    int   tI[TOPK];
#pragma unroll
    for (int j = 0; j < TOPK; ++j) { tS[j] = -INFINITY; tI[j] = 0x7fffffff; }

    // ISA 16-bit fragment lane layouts (CDNA5 ISA):
    // A 16x32: lane<16 -> row=lane,  K={0..7,16..23};  lane>=16 -> K={8..15,24..31}
    // B 32x16: lane<16 -> col=lane,  K={0..7,16..23};  lane>=16 -> K={8..15,24..31}
    const int aRowB = (rowBase + ln15) * 80 + (lhi ? 16 : 0);
    const int bOffK = lhi * 16;

    for (int tt = 0; tt < TILES; ++tt) {
        const int mBase = mch * MCHUNK + tt * MT;
        v8f acc[2][4];
        const v8f vz = {0.f, 0.f, 0.f, 0.f, 0.f, 0.f, 0.f, 0.f};
#pragma unroll
        for (int a = 0; a < 2; ++a)
#pragma unroll
            for (int b = 0; b < 4; ++b) acc[a][b] = vz;
        float nacc[4] = {0.f, 0.f, 0.f, 0.f};

        for (int ks = 0; ks < DIM / KT; ++ks) {
            // Stage A tile: 128x32 f16 from qn (L2-resident), coalesced 16B.
#pragma unroll
            for (int q = 0; q < 2; ++q) {
                const int id = t + 256 * q;            // 0..511
                const int r  = id >> 2;
                const int c  = id & 3;
                uint4 av = *(const uint4*)((const unsigned char*)qn +
                            (size_t)(bRow0 + r) * (DIM * 2) + ks * 64 + c * 16);
                *(uint4*)((unsigned char*)lA + r * 80 + c * 16) = av;
            }
            // Stage B tile: 128x32 f32 keys -> f16, inline sum-of-squares.
#pragma unroll
            for (int q = 0; q < 4; ++q) {
                const int id = t + 256 * q;            // 0..1023
                const int r  = id >> 3;
                const int c4 = id & 7;
                float4 kv = *(const float4*)(keys +
                            (size_t)(mBase + r) * DIM + ks * KT + c4 * 4);
                nacc[q] = fmaf(kv.x, kv.x, nacc[q]);
                nacc[q] = fmaf(kv.y, kv.y, nacc[q]);
                nacc[q] = fmaf(kv.z, kv.z, nacc[q]);
                nacc[q] = fmaf(kv.w, kv.w, nacc[q]);
                v4h pk;
                pk.x = (f16_t)kv.x; pk.y = (f16_t)kv.y;
                pk.z = (f16_t)kv.z; pk.w = (f16_t)kv.w;
                *(v4h*)((unsigned char*)lB + r * 80 + c4 * 8) = pk;
            }
            __syncthreads();

            v16h A0, A1;
            {
                union { v16h v; uint4 q[2]; } u;
                const unsigned char* p = (const unsigned char*)lA + aRowB;
                u.q[0] = *(const uint4*)p;
                u.q[1] = *(const uint4*)(p + 32);
                A0 = u.v;
                p += 16 * 80;
                u.q[0] = *(const uint4*)p;
                u.q[1] = *(const uint4*)(p + 32);
                A1 = u.v;
            }
#pragma unroll
            for (int bf = 0; bf < 4; ++bf) {
                union { v16h v; uint4 q[2]; } u;
                const unsigned char* p = (const unsigned char*)lB +
                        (colBase + 16 * bf + ln15) * 80 + bOffK;
                u.q[0] = *(const uint4*)p;
                u.q[1] = *(const uint4*)(p + 32);
                const v16h Bf = u.v;
                acc[0][bf] = __builtin_amdgcn_wmma_f32_16x16x32_f16(
                        false, A0, false, Bf, (short)0, acc[0][bf], false, false);
                acc[1][bf] = __builtin_amdgcn_wmma_f32_16x16x32_f16(
                        false, A1, false, Bf, (short)0, acc[1][bf], false, false);
            }
            __syncthreads();
        }

        // Finalize key-row inverse norms (8 consecutive lanes share a row).
#pragma unroll
        for (int q = 0; q < 4; ++q) {
            float v = nacc[q];
            v += __shfl_xor(v, 1);
            v += __shfl_xor(v, 2);
            v += __shfl_xor(v, 4);
            if ((t & 7) == 0) lRnorm[(t >> 3) + 32 * q] = v;
        }
        __syncthreads();
        if (t < MT) lRnorm[t] = 1.0f / fmaxf(sqrtf(lRnorm[t]), 1e-12f);
        __syncthreads();

        // Two 64-col rounds: dump scaled scores, then guarded top-32 insert.
        for (int rr = 0; rr < 2; ++rr) {
            if (wc == rr) {
#pragma unroll
                for (int a = 0; a < 2; ++a)
#pragma unroll
                    for (int bf = 0; bf < 4; ++bf) {
                        const int col = 16 * bf + ln15;          // 0..63
                        const float rn = lRnorm[colBase + col];
#pragma unroll
                        for (int v = 0; v < 8; ++v) {
                            const int row = rowBase + 16 * a + v + 8 * lhi;
                            lScore[row * 65 + col] = acc[a][bf][v] * rn;
                        }
                    }
            }
            __syncthreads();
            {
                const int r   = t & 127;
                const int h   = t >> 7;
                const int gm0 = mBase + rr * 64 + h * 32;
#pragma unroll 1
                for (int j = 0; j < 32; ++j) {
                    const float s  = lScore[r * 65 + h * 32 + j];
                    const int   gi = gm0 + j;
                    if (better(s, gi, tS[TOPK - 1], tI[TOPK - 1])) {
                        float cs = s; int ci = gi;
#pragma unroll
                        for (int p = 0; p < TOPK; ++p) {
                            const bool bt = better(cs, ci, tS[p], tI[p]);
                            const float os = tS[p]; const int oi = tI[p];
                            if (bt) { tS[p] = cs; tI[p] = ci; cs = os; ci = oi; }
                        }
                    }
                }
            }
            __syncthreads();
        }
    }

    // Merge the two per-row half-lists, emit chunk candidates.
    if (t >= 128) {
        const int r = t - 128;
#pragma unroll
        for (int j = 0; j < TOPK; ++j) { mS[r * 32 + j] = tS[j]; mI[r * 32 + j] = tI[j]; }
    }
    __syncthreads();
    if (t < 128) {
#pragma unroll 1
        for (int j = 0; j < TOPK; ++j) {   // partner list is sorted -> early out
            const float s  = mS[t * 32 + j];
            const int   gi = mI[t * 32 + j];
            if (!better(s, gi, tS[TOPK - 1], tI[TOPK - 1])) break;
            float cs = s; int ci = gi;
#pragma unroll
            for (int p = 0; p < TOPK; ++p) {
                const bool bt = better(cs, ci, tS[p], tI[p]);
                const float os = tS[p]; const int oi = tI[p];
                if (bt) { tS[p] = cs; tI[p] = ci; cs = os; ci = oi; }
            }
        }
        const int row = bRow0 + t;
        const size_t base = ((size_t)row * NMCH + mch) * TOPK;
#pragma unroll
        for (int j = 0; j < TOPK; ++j) { candS[base + j] = tS[j]; candI[base + j] = tI[j]; }
    }
}

// ---------------------------------------------------------------------------
// K3: per-row exact merge of NMCH*32 candidates -> top-32, gather+sum values.
// ---------------------------------------------------------------------------
__global__ __launch_bounds__(256) void k3_merge_gather(
        const float* __restrict__ candS, const int* __restrict__ candI,
        const float* __restrict__ values, float* __restrict__ out) {
    __shared__ unsigned long long cK[NMCH * TOPK];   // 4096 packed keys (32 KB)
    __shared__ unsigned long long rK[256];
    __shared__ int rP[256];
    __shared__ int sel[TOPK];
    const int row = blockIdx.x;
    const int t   = threadIdx.x;
#pragma unroll
    for (int q = 0; q < (NMCH * TOPK) / 256; ++q) {
        const int p = t + 256 * q;
        const float s = candS[(size_t)row * (NMCH * TOPK) + p];
        const int   i = candI[(size_t)row * (NMCH * TOPK) + p];
        const unsigned u = __float_as_uint(s);
        const unsigned m = (u & 0x80000000u) ? ~u : (u | 0x80000000u);
        cK[p] = ((unsigned long long)m << 32) |
                (unsigned long long)(0xffffffffu - (unsigned)i);
    }
    __syncthreads();
    for (int round = 0; round < TOPK; ++round) {
        unsigned long long best = 0ull; int bp = 0;
#pragma unroll
        for (int q = 0; q < (NMCH * TOPK) / 256; ++q) {
            const int p = t + 256 * q;
            const unsigned long long k = cK[p];
            if (k > best) { best = k; bp = p; }
        }
        rK[t] = best; rP[t] = bp;
        __syncthreads();
        for (int off = 128; off > 0; off >>= 1) {
            if (t < off && rK[t + off] > rK[t]) { rK[t] = rK[t + off]; rP[t] = rP[t + off]; }
            __syncthreads();
        }
        if (t == 0) {
            sel[round] = (int)(0xffffffffu - (unsigned)(rK[0] & 0xffffffffull));
            cK[rP[0]]  = 0ull;
        }
        __syncthreads();
    }
    float a0 = 0.f, a1 = 0.f;
#pragma unroll 1
    for (int s = 0; s < TOPK; ++s) {
        const int i = sel[s];
        a0 += values[(size_t)i * DIM + t];
        a1 += values[(size_t)i * DIM + t + 256];
    }
    out[(size_t)row * DIM + t]       = a0;
    out[(size_t)row * DIM + t + 256] = a1;
}

// ---------------------------------------------------------------------------
extern "C" void kernel_launch(void* const* d_in, const int* in_sizes, int n_in,
                              void* d_out, int out_size, void* d_ws, size_t ws_size,
                              hipStream_t stream) {
    const float* x      = (const float*)d_in[0];
    const float* Wq     = (const float*)d_in[1];
    const float* bq     = (const float*)d_in[2];
    const float* keys   = (const float*)d_in[3];
    const float* values = (const float*)d_in[4];
    float* out = (float*)d_out;

    unsigned char* ws = (unsigned char*)d_ws;
    f16_t* qn    = (f16_t*)ws;                                  // 512*512*2    = 0.5 MB
    float* candS = (float*)(ws + (size_t)BATCH * DIM * 2);      // 512*128*32*4 = 8 MB
    int*   candI = (int*)(ws + (size_t)BATCH * DIM * 2 +
                          (size_t)BATCH * NMCH * TOPK * 4);     // 8 MB

    k1_project_norm<<<BATCH, 256, 0, stream>>>(x, Wq, bq, qn);
    dim3 g2(NBCH, NMCH);   // x fastest: 4 batch-chunks share each key chunk in L2
    k2_score_topk<<<g2, 256, 0, stream>>>(qn, keys, candS, candI);
    k3_merge_gather<<<BATCH, 256, 0, stream>>>(candS, candI, values, out);
}
